// GameOfLifeTransformer_39316130628012
// MI455X (gfx1250) — matrix operand order repaired and checked by reading, not compiled, not run on hardware
//
#include <hip/hip_runtime.h>
#include <hip/hip_bf16.h>

typedef __attribute__((ext_vector_type(16))) _Float16 v16h;
typedef __attribute__((ext_vector_type(8)))  _Float16 v8h;
typedef __attribute__((ext_vector_type(8)))  float    v8f;

#define HH 48
#define WW 48
#define LL (HH*WW)        // 2304
#define NHD 8             // heads
#define HDM 32            // head dim
#define DM 256            // d_model
#define MLPD 1024
#define NLAYER 4
#define BBATCH 4
#define BLD (BBATCH*LL)   // 9216 rows
#define NBUCKET 15

// ---------------------------------------------------------------------------
// K-swizzle for f16 activation buffers that feed WMMA as the A matrix.
// Within each 32-wide K block, elements are permuted so that one lane-half's
// 16 A-fragment halves are contiguous:
//   pos = base32 + hf(j)*16 + hi(j)*8 + (j&7),  hf=(j>>3)&1, hi=(j>>4)&1
// The GEMM then loads each A fragment as a single 32-byte b128.
// ---------------------------------------------------------------------------
__device__ __forceinline__ int swzK(int d) {
    int j = d & 31;
    return (d & ~31) + (((j >> 3) & 1) << 4) + (((j >> 4) & 1) << 3) + (j & 7);
}

// ---------------------------------------------------------------------------
// fp32 -> fp16 conversion (weights; natural layout, B side of WMMA)
// ---------------------------------------------------------------------------
__global__ void f32_to_f16_kernel(const float* __restrict__ src,
                                  _Float16* __restrict__ dst, int n) {
    int i = blockIdx.x * blockDim.x + threadIdx.x;
    if (i < n) dst[i] = (_Float16)src[i];
}

// ---------------------------------------------------------------------------
// Embedding: h[b,l,d] = x*Win[0,d] + (y/47)*Win[1,d] + (x/47)*Win[2,d] + b_in
// ---------------------------------------------------------------------------
__global__ void embed_kernel(const int* __restrict__ x,
                             const float* __restrict__ Win,
                             const float* __restrict__ b_in,
                             float* __restrict__ h) {
    int idx = blockIdx.x * blockDim.x + threadIdx.x;   // BLD*DM threads
    int d  = idx & (DM - 1);
    int bl = idx >> 8;
    int b  = bl / LL;
    int l  = bl % LL;
    int yy = l / WW, xx = l % WW;
    float f0 = (float)x[b * LL + l];
    float f1 = (float)yy * (1.0f / 47.0f);
    float f2 = (float)xx * (1.0f / 47.0f);
    h[idx] = f0 * Win[d] + f1 * Win[DM + d] + f2 * Win[2 * DM + d] + b_in[d];
}

// ---------------------------------------------------------------------------
// LayerNorm over D=256, f16 output written K-swizzled (feeds WMMA A side).
// One block (256 threads) per row.
// ---------------------------------------------------------------------------
__global__ void ln_f16_kernel(const float* __restrict__ X,
                              const float* __restrict__ s,
                              const float* __restrict__ bta,
                              _Float16* __restrict__ Y) {
    __shared__ float sb[16];
    int row = blockIdx.x;
    int d = threadIdx.x;                     // 256 threads
    float xv = X[(size_t)row * DM + d];
    float v = xv, v2 = xv * xv;
    for (int off = 16; off; off >>= 1) {
        v  += __shfl_xor(v,  off, 32);
        v2 += __shfl_xor(v2, off, 32);
    }
    int w = threadIdx.x >> 5;
    if ((threadIdx.x & 31) == 0) { sb[w] = v; sb[8 + w] = v2; }
    __syncthreads();
    if (threadIdx.x == 0) {
        float a = 0.f, b2 = 0.f;
        for (int i = 0; i < 8; ++i) { a += sb[i]; b2 += sb[8 + i]; }
        sb[0] = a; sb[8] = b2;
    }
    __syncthreads();
    float mean = sb[0] * (1.0f / DM);
    float var  = sb[8] * (1.0f / DM) - mean * mean;
    float r = rsqrtf(var + 1e-6f);
    Y[(size_t)row * DM + swzK(d)] = (_Float16)((xv - mean) * r * s[d] + bta[d]);
}

// ---------------------------------------------------------------------------
// WMMA GEMM: C[M,N] = A[M,K](f16, K-swizzled) x Bw[K,N](f16) + bias[N]
// Register-blocked: one wave computes a 32(M) x 64(N) macro-tile as
// 2 x 4 fragments -> 8 v_wmma_f32_16x16x32_f16 per K-step vs 2 A v16h loads + 64 scalar B f16 loads.
// K-loop kept at unroll 1 so live set (~120 VGPRs) never spills.
//   mode 0: Cf = acc + bias                          (f32 out, natural)
//   mode 1: Cf = acc + bias + residual (in-place h)  (f32 out, natural)
//   mode 2: Ch = gelu_tanh(acc + bias)               (f16 out, K-swizzled)
// ---------------------------------------------------------------------------
__global__ void wmma_gemm_kernel(const _Float16* __restrict__ A,
                                 const _Float16* __restrict__ Bw,
                                 const float* __restrict__ bias,
                                 const float* __restrict__ residual,
                                 float* __restrict__ Cf,
                                 _Float16* __restrict__ Ch,
                                 int M, int K, int N, int mode) {
    int wave   = blockIdx.x * (blockDim.x >> 5) + (threadIdx.x >> 5);
    int tilesN = N >> 6;                      // 64-wide N macro-tiles
    int tM = wave / tilesN;                   // 32-row M macro-tile
    int tN = wave % tilesN;
    if (tM * 32 >= M) return;                 // wave-uniform; EXEC stays full
    int lane = threadIdx.x & 31;
    int hf   = lane >> 4;                     // lane half selects K sub-block
    int m0   = tM * 32 + (lane & 15);         // A: lane%16 = M row
    int m1   = m0 + 16;

    const _Float16* arow0 = A + (size_t)m0 * K + hf * 16;
    const _Float16* arow1 = A + (size_t)m1 * K + hf * 16;
    const _Float16* bbase = Bw + tN * 64;

    v8f acc[2][4] = {};
    #pragma unroll 1
    for (int kk = 0; kk < K; kk += 32) {
        // A is stored K-swizzled: one contiguous b128 per fragment per lane.
        v16h a0 = *(const v16h*)(arow0 + kk);
        v16h a1 = *(const v16h*)(arow1 + kk);
        // B 32x16 layout: lane%16 = column N, element s = K row
        // k = 16*(s>>3) + 8*hf + (s&7) (matches swizzled A).
        const _Float16* brow = bbase + (lane & 15);
        v16h b0, b1, b2, b3;
        #pragma unroll
        for (int s = 0; s < 16; ++s) {
            const _Float16* p = brow + (size_t)(kk + ((s >> 3) << 4) + hf * 8 + (s & 7)) * N;
            b0[s] = p[0];
            b1[s] = p[16];
            b2[s] = p[32];
            b3[s] = p[48];
        }

        acc[0][0] = __builtin_amdgcn_wmma_f32_16x16x32_f16(
            false, a0, false, b0, (short)0, acc[0][0], false, false);
        acc[0][1] = __builtin_amdgcn_wmma_f32_16x16x32_f16(
            false, a0, false, b1, (short)0, acc[0][1], false, false);
        acc[0][2] = __builtin_amdgcn_wmma_f32_16x16x32_f16(
            false, a0, false, b2, (short)0, acc[0][2], false, false);
        acc[0][3] = __builtin_amdgcn_wmma_f32_16x16x32_f16(
            false, a0, false, b3, (short)0, acc[0][3], false, false);
        acc[1][0] = __builtin_amdgcn_wmma_f32_16x16x32_f16(
            false, a1, false, b0, (short)0, acc[1][0], false, false);
        acc[1][1] = __builtin_amdgcn_wmma_f32_16x16x32_f16(
            false, a1, false, b1, (short)0, acc[1][1], false, false);
        acc[1][2] = __builtin_amdgcn_wmma_f32_16x16x32_f16(
            false, a1, false, b2, (short)0, acc[1][2], false, false);
        acc[1][3] = __builtin_amdgcn_wmma_f32_16x16x32_f16(
            false, a1, false, b3, (short)0, acc[1][3], false, false);
    }

    // C/D layout: lane%16 = N col, VGPR r -> M = r + 8*(lane/16)
    #pragma unroll
    for (int mi = 0; mi < 2; ++mi) {
        int rbase = tM * 32 + mi * 16 + hf * 8;
        #pragma unroll
        for (int ni = 0; ni < 4; ++ni) {
            int col = tN * 64 + ni * 16 + (lane & 15);
            float bcol = bias[col];
            #pragma unroll
            for (int r = 0; r < 8; ++r) {
                float val = acc[mi][ni][r] + bcol;
                if (mode == 1) {
                    size_t idx = (size_t)(rbase + r) * N + col;
                    Cf[idx] = val + residual[idx];
                } else if (mode == 2) {
                    float t = 0.7978845608028654f *
                              (val + 0.044715f * val * val * val);
                    // output feeds next GEMM's A side -> store K-swizzled
                    size_t idx = (size_t)(rbase + r) * N + swzK(col);
                    Ch[idx] = (_Float16)(0.5f * val * (1.0f + tanhf(t)));
                } else {
                    size_t idx = (size_t)(rbase + r) * N + col;
                    Cf[idx] = val;
                }
            }
        }
    }
}

// ---------------------------------------------------------------------------
// Local windowed attention (radius 2, periodic) + learned rel-pos bias.
// One wave per (b, head, query); lane = head-dim element.
// Off-window keys carry -1e9 mask in the reference -> contribute exactly 0,
// so only the 25 in-window keys are computed. Output f16, K-swizzled
// (feeds the Wo WMMA GEMM as A).
// ---------------------------------------------------------------------------
__global__ void attn_local_kernel(const float* __restrict__ q,
                                  const float* __restrict__ k,
                                  const float* __restrict__ v,
                                  const float* __restrict__ rel,
                                  _Float16* __restrict__ att16) {
    int wave = blockIdx.x * (blockDim.x >> 5) + (threadIdx.x >> 5);
    int lane = threadIdx.x & 31;
    int ql  = wave % LL;
    int tmp = wave / LL;
    int hh  = tmp % NHD;
    int b   = tmp / NHD;
    int qy = ql / WW, qx = ql % WW;
    const float scale = 0.17677669529663687f;   // 1/sqrt(32)

    float qv = q[((size_t)(b * LL + ql)) * DM + hh * HDM + lane];
    float sc[25];
    #pragma unroll
    for (int idx = 0; idx < 25; ++idx) {
        int dy = idx / 5 - 2, dx = idx % 5 - 2;
        int ky = qy + dy; ky += (ky < 0) ? HH : 0; ky -= (ky >= HH) ? HH : 0;
        int kx = qx + dx; kx += (kx < 0) ? WW : 0; kx -= (kx >= WW) ? WW : 0;
        int kl = ky * WW + kx;
        float s = qv * k[((size_t)(b * LL + kl)) * DM + hh * HDM + lane];
        for (int off = 16; off; off >>= 1) s += __shfl_xor(s, off, 32);
        sc[idx] = s * scale +
                  rel[hh * (NBUCKET * NBUCKET) + (dy + 7) * NBUCKET + (dx + 7)];
    }
    float mx = sc[0];
    #pragma unroll
    for (int i = 1; i < 25; ++i) mx = fmaxf(mx, sc[i]);
    float sum = 0.f;
    #pragma unroll
    for (int i = 0; i < 25; ++i) { sc[i] = __expf(sc[i] - mx); sum += sc[i]; }
    float inv = 1.0f / sum;
    float out = 0.f;
    #pragma unroll
    for (int idx = 0; idx < 25; ++idx) {
        int dy = idx / 5 - 2, dx = idx % 5 - 2;
        int ky = qy + dy; ky += (ky < 0) ? HH : 0; ky -= (ky >= HH) ? HH : 0;
        int kx = qx + dx; kx += (kx < 0) ? WW : 0; kx -= (kx >= WW) ? WW : 0;
        int kl = ky * WW + kx;
        out += sc[idx] * v[((size_t)(b * LL + kl)) * DM + hh * HDM + lane];
    }
    att16[((size_t)(b * LL + ql)) * DM + swzK(hh * HDM + lane)] =
        (_Float16)(out * inv);
}

// ---------------------------------------------------------------------------
// Final LN + D->1 projection. One block per row.
// ---------------------------------------------------------------------------
__global__ void final_ln_proj_kernel(const float* __restrict__ X,
                                     const float* __restrict__ s,
                                     const float* __restrict__ bta,
                                     const float* __restrict__ Wout,
                                     const float* __restrict__ bout,
                                     float* __restrict__ out) {
    __shared__ float sb[24];
    int row = blockIdx.x;
    int d = threadIdx.x;
    float xv = X[(size_t)row * DM + d];
    float v = xv, v2 = xv * xv;
    for (int off = 16; off; off >>= 1) {
        v  += __shfl_xor(v,  off, 32);
        v2 += __shfl_xor(v2, off, 32);
    }
    int w = threadIdx.x >> 5;
    if ((threadIdx.x & 31) == 0) { sb[w] = v; sb[8 + w] = v2; }
    __syncthreads();
    if (threadIdx.x == 0) {
        float a = 0.f, b2 = 0.f;
        for (int i = 0; i < 8; ++i) { a += sb[i]; b2 += sb[8 + i]; }
        sb[0] = a; sb[8] = b2;
    }
    __syncthreads();
    float mean = sb[0] * (1.0f / DM);
    float var  = sb[8] * (1.0f / DM) - mean * mean;
    float r = rsqrtf(var + 1e-6f);
    float yn = (xv - mean) * r * s[d] + bta[d];
    float t = yn * Wout[d];
    for (int off = 16; off; off >>= 1) t += __shfl_xor(t, off, 32);
    if ((threadIdx.x & 31) == 0) sb[16 + w] = t;
    __syncthreads();
    if (threadIdx.x == 0) {
        float a = 0.f;
        for (int i = 0; i < 8; ++i) a += sb[16 + i];
        out[row] = a + bout[0];
    }
}

// ---------------------------------------------------------------------------
extern "C" void kernel_launch(void* const* d_in, const int* in_sizes, int n_in,
                              void* d_out, int out_size, void* d_ws, size_t ws_size,
                              hipStream_t stream) {
    (void)in_sizes; (void)n_in; (void)out_size; (void)ws_size;

    const int*   x     = (const int*)  d_in[0];
    const float* Win   = (const float*)d_in[1];
    const float* b_in  = (const float*)d_in[2];
    const float* lnA_s = (const float*)d_in[3];
    const float* lnA_b = (const float*)d_in[4];
    const float* Wq    = (const float*)d_in[5];
    const float* bq    = (const float*)d_in[6];
    const float* Wk    = (const float*)d_in[7];
    const float* bk    = (const float*)d_in[8];
    const float* Wv    = (const float*)d_in[9];
    const float* bv    = (const float*)d_in[10];
    const float* Wo    = (const float*)d_in[11];
    const float* bo    = (const float*)d_in[12];
    const float* lnB_s = (const float*)d_in[13];
    const float* lnB_b = (const float*)d_in[14];
    const float* W1    = (const float*)d_in[15];
    const float* b1    = (const float*)d_in[16];
    const float* W2    = (const float*)d_in[17];
    const float* b2    = (const float*)d_in[18];
    const float* rel   = (const float*)d_in[19];
    const float* lnF_s = (const float*)d_in[20];
    const float* lnF_b = (const float*)d_in[21];
    const float* Woutp = (const float*)d_in[22];
    const float* boutp = (const float*)d_in[23];

    char* ws = (char*)d_ws;
    size_t off = 0;
    auto carve = [&](size_t bytes) -> char* {
        char* p = ws + off;
        off += (bytes + 255) & ~(size_t)255;
        return p;
    };
    float*    h     = (float*)   carve((size_t)BLD * DM * 4);
    _Float16* y16   = (_Float16*)carve((size_t)BLD * DM * 2);
    float*    qb    = (float*)   carve((size_t)BLD * DM * 4);
    float*    kb    = (float*)   carve((size_t)BLD * DM * 4);
    float*    vb    = (float*)   carve((size_t)BLD * DM * 4);
    _Float16* att16 = (_Float16*)carve((size_t)BLD * DM * 2);
    _Float16* mid16 = (_Float16*)carve((size_t)BLD * MLPD * 2);
    _Float16* wq16  = (_Float16*)carve((size_t)NLAYER * DM * DM * 2);
    _Float16* wk16  = (_Float16*)carve((size_t)NLAYER * DM * DM * 2);
    _Float16* wv16  = (_Float16*)carve((size_t)NLAYER * DM * DM * 2);
    _Float16* wo16  = (_Float16*)carve((size_t)NLAYER * DM * DM * 2);
    _Float16* w116  = (_Float16*)carve((size_t)NLAYER * DM * MLPD * 2);
    _Float16* w216  = (_Float16*)carve((size_t)NLAYER * MLPD * DM * 2);

    // Convert weights fp32 -> fp16 (every call; deterministic, no caching).
    const int nWd = NLAYER * DM * DM;
    const int nWm = NLAYER * DM * MLPD;
    f32_to_f16_kernel<<<(nWd + 255) / 256, 256, 0, stream>>>(Wq, wq16, nWd);
    f32_to_f16_kernel<<<(nWd + 255) / 256, 256, 0, stream>>>(Wk, wk16, nWd);
    f32_to_f16_kernel<<<(nWd + 255) / 256, 256, 0, stream>>>(Wv, wv16, nWd);
    f32_to_f16_kernel<<<(nWd + 255) / 256, 256, 0, stream>>>(Wo, wo16, nWd);
    f32_to_f16_kernel<<<(nWm + 255) / 256, 256, 0, stream>>>(W1, w116, nWm);
    f32_to_f16_kernel<<<(nWm + 255) / 256, 256, 0, stream>>>(W2, w216, nWm);

    embed_kernel<<<(BLD * DM) / 256, 256, 0, stream>>>(x, Win, b_in, h);

    // Macro-tile grids: waves = (M/32) * (N/64); 8 waves per block.
    const int gDD  = (BLD / 32) * (DM / 64) / 8;    // 144 blocks (N=256 GEMMs)
    const int gDM  = (BLD / 32) * (MLPD / 64) / 8;  // 576 blocks (N=1024 GEMM)
    const int gATT = (BBATCH * NHD * LL) / 8;       // 9216 blocks

    for (int i = 0; i < NLAYER; ++i) {
        ln_f16_kernel<<<BLD, 256, 0, stream>>>(h, lnA_s + i * DM, lnA_b + i * DM, y16);
        wmma_gemm_kernel<<<gDD, 256, 0, stream>>>(
            y16, wq16 + (size_t)i * DM * DM, bq + i * DM, nullptr, qb, nullptr,
            BLD, DM, DM, 0);
        wmma_gemm_kernel<<<gDD, 256, 0, stream>>>(
            y16, wk16 + (size_t)i * DM * DM, bk + i * DM, nullptr, kb, nullptr,
            BLD, DM, DM, 0);
        wmma_gemm_kernel<<<gDD, 256, 0, stream>>>(
            y16, wv16 + (size_t)i * DM * DM, bv + i * DM, nullptr, vb, nullptr,
            BLD, DM, DM, 0);
        attn_local_kernel<<<gATT, 256, 0, stream>>>(qb, kb, vb, rel, att16);
        wmma_gemm_kernel<<<gDD, 256, 0, stream>>>(
            att16, wo16 + (size_t)i * DM * DM, bo + i * DM, h, h, nullptr,
            BLD, DM, DM, 1);
        ln_f16_kernel<<<BLD, 256, 0, stream>>>(h, lnB_s + i * DM, lnB_b + i * DM, y16);
        wmma_gemm_kernel<<<gDM, 256, 0, stream>>>(
            y16, w116 + (size_t)i * DM * MLPD, b1 + i * MLPD, nullptr, nullptr,
            mid16, BLD, DM, MLPD, 2);
        wmma_gemm_kernel<<<gDD, 256, 0, stream>>>(
            mid16, w216 + (size_t)i * MLPD * DM, b2 + i * DM, h, h, nullptr,
            BLD, MLPD, DM, 1);
    }

    final_ln_proj_kernel<<<BLD, 256, 0, stream>>>(
        h, lnF_s, lnF_b, Woutp, boutp, (float*)d_out);
}
